// TriangleAttention_52158082842752
// MI455X (gfx1250) — matrix operand order repaired and checked by reading, compiled, not run on hardware
//
#include <hip/hip_runtime.h>
#include <hip/hip_bf16.h>
#include <math.h>

typedef __attribute__((ext_vector_type(16))) _Float16 v16h;
typedef __attribute__((ext_vector_type(8)))  _Float16 v8h;
typedef __attribute__((ext_vector_type(8)))  float    v8f;
typedef _Float16 h16;
typedef unsigned int u32;

// Problem constants (from reference): B=2, L=256, D=64, H=4, Dh=16
constexpr int    LSEQ = 256;
constexpr int    NB   = 2;
constexpr size_t RD   = (size_t)NB * LSEQ * LSEQ * 64;  // 8,388,608 elements

// ---------------- WMMA operand builders (CDNA5 16-bit layouts) ----------------
// A operand (16x32 MxK): lane m=lane&15; element e <-> k = e + 16*hi
__device__ __forceinline__ v16h mkA(const h16* __restrict__ row, int k0, int hi, int kmax) {
  v16h a;
#pragma unroll
  for (int e = 0; e < 16; ++e) {
    int k = k0 + e + 16 * hi;
    a[e] = (k < kmax) ? row[k] : (h16)0.0f;
  }
  return a;
}
// B operand (32x16 KxN) from an [N][K] row-major source (e.g. weight W[n][k]):
// lane n=lane&15; element e <-> k = e + 16*hi  (contiguous per lane -> b128 loads)
__device__ __forceinline__ v16h mkB_NT(const h16* __restrict__ srcNT, int ld, int n,
                                       int k0, int hi, int kmax) {
  v16h b;
#pragma unroll
  for (int e = 0; e < 16; ++e) {
    int k = k0 + e + 16 * hi;
    b[e] = (k < kmax) ? srcNT[n * ld + k] : (h16)0.0f;
  }
  return b;
}
__device__ __forceinline__ v8f wmma16(v16h a, v16h b, v8f c) {
  return __builtin_amdgcn_wmma_f32_16x16x32_f16(false, a, false, b, (short)0, c, false, false);
}

// ---------------- K0: weight fp32 -> fp16 ----------------
__global__ void k_cvt_f16(const float* __restrict__ src, h16* __restrict__ dst, int n) {
  int i = blockIdx.x * blockDim.x + threadIdx.x;
  if (i < n) dst[i] = (h16)src[i];
}

// ---------------- K1: fused LayerNorm + Q/K/V/G projections ----------------
// block = 256 threads (8 waves), 32 rows per block.
// Each wave owns one (rowtile, n0) pair and iterates the 4 weight matrices
// with a compile-time-constant `mat` (no divergent epilogue branches).
__global__ __launch_bounds__(256) void k_ln_qkvg(
    const float* __restrict__ z, const float* __restrict__ gamma, const float* __restrict__ beta,
    const h16* __restrict__ Wh /* QKVG contiguous, 4*64*64 f16 */, const float* __restrict__ bg,
    h16* __restrict__ Qh, h16* __restrict__ Kh, h16* __restrict__ Vh, h16* __restrict__ Gh) {
  __shared__ h16 zsh[32 * 64];        // 4 KB  normalized activations (f16)
  __shared__ h16 Wsh[4 * 64 * 64];    // 32 KB Wq,Wk,Wv,Wg
  int t = threadIdx.x;
  size_t row0 = (size_t)blockIdx.x * 32;

  // stage weights into LDS (f16, dword copies)
  for (int i = t; i < 4 * 64 * 64 / 2; i += 256)
    ((u32*)Wsh)[i] = ((const u32*)Wh)[i];

  // LayerNorm: 8 lanes per row, 8 elements per lane
  int rl = t >> 3, cq = (t & 7) * 8;
  const float* zp = z + (row0 + rl) * 64 + cq;
  float x[8], s = 0.f, ss = 0.f;
#pragma unroll
  for (int i = 0; i < 8; ++i) { x[i] = zp[i]; s += x[i]; ss += x[i] * x[i]; }
  s  += __shfl_xor(s, 1);  s  += __shfl_xor(s, 2);  s  += __shfl_xor(s, 4);
  ss += __shfl_xor(ss, 1); ss += __shfl_xor(ss, 2); ss += __shfl_xor(ss, 4);
  float mu  = s * (1.0f / 64.0f);
  float var = ss * (1.0f / 64.0f) - mu * mu;
  float inv = rsqrtf(var + 1e-5f);
#pragma unroll
  for (int i = 0; i < 8; ++i)
    zsh[rl * 64 + cq + i] = (h16)((x[i] - mu) * inv * gamma[cq + i] + beta[cq + i]);
  __syncthreads();

  int w = t >> 5, lane = t & 31, n = lane & 15, hi = lane >> 4;
  int rowtile = w >> 2;          // 0..1
  int n0 = (w & 3) * 16;         // 0..48
  const h16* arow = zsh + (rowtile * 16 + (lane & 15)) * 64;
  v16h a0 = mkA(arow, 0, hi, 64);
  v16h a1 = mkA(arow, 32, hi, 64);
  float bgv = bg[n0 + n];
#pragma unroll
  for (int mat = 0; mat < 4; ++mat) {   // compile-time constant after unroll
    const h16* wm = Wsh + mat * 4096;
    v8f c = {};
    c = wmma16(a0, mkB_NT(wm, 64, n0 + n, 0, hi, 64), c);
    c = wmma16(a1, mkB_NT(wm, 64, n0 + n, 32, hi, 64), c);
#pragma unroll
    for (int r = 0; r < 8; ++r) {
      int m = r + 8 * hi;
      size_t idx = (row0 + rowtile * 16 + m) * 64 + n0 + n;
      float v = c[r];
      if (mat == 0)      Qh[idx] = (h16)(v * 0.25f);  // fold 1/sqrt(Dh)
      else if (mat == 1) Kh[idx] = (h16)v;
      else if (mat == 2) Vh[idx] = (h16)v;
      else Gh[idx] = (h16)__builtin_amdgcn_rcpf(1.0f + __expf(-(v + bgv)));
    }
  }
}

// ---------------- K2: per-(b,row,head) attention + softmax + gate ----------------
// grid = (L*H, B), block = 128 threads (4 waves). Each wave owns 64 query rows.
__global__ __launch_bounds__(128) void k_attn(
    const h16* __restrict__ Qh, const h16* __restrict__ Kh, const h16* __restrict__ Vh,
    const h16* __restrict__ Gh, h16* __restrict__ Ph) {
  __shared__ h16 Qs[256 * 16];       // [j][d]  8 KB
  __shared__ h16 Ks[256 * 16];       // [j][d]  8 KB
  __shared__ h16 Vt[16 * 256];       // [d][j]  8 KB  (transposed for contiguous B gather)
  __shared__ h16 Ps[4][16 * 256];    // 32 KB (per-wave P)
  __shared__ float rsc[4][16];
  int t = threadIdx.x;
  int h = blockIdx.x & 3, l = blockIdx.x >> 2;
  size_t base = ((size_t)blockIdx.y * LSEQ + l) * LSEQ;

  // warm the gate rows we will gather in the epilogue (global_prefetch_b8)
  for (int r = t; r < 256; r += 128)
    __builtin_prefetch(Gh + (base + r) * 64 + h * 16, 0, 3);

  // stage this head's Q/K (row-major) and V (transposed) into LDS
  for (int i = t; i < 2048; i += 128) {
    int row = i >> 3, c = i & 7;
    size_t goff = (base + row) * 64 + h * 16;
    ((u32*)Qs)[row * 8 + c] = ((const u32*)(Qh + goff))[c];
    ((u32*)Ks)[row * 8 + c] = ((const u32*)(Kh + goff))[c];
    union { u32 u; h16 f[2]; } vv;
    vv.u = ((const u32*)(Vh + goff))[c];
    Vt[(2 * c + 0) * 256 + row] = vv.f[0];
    Vt[(2 * c + 1) * 256 + row] = vv.f[1];
  }
  __syncthreads();

  int w = t >> 5, lane = t & 31, n = lane & 15, hi = lane >> 4;
  h16* Pw = Ps[w];
  for (int it = 0; it < 4; ++it) {
    int i0 = w * 64 + it * 16;
    // S = Q K^T  (K-dim = Dh = 16, zero-padded to 32)
    v16h aq = mkA(Qs + (i0 + (lane & 15)) * 16, 0, hi, 16);
#pragma unroll
    for (int jt = 0; jt < 16; ++jt) {
      v8f s = {};
      s = wmma16(aq, mkB_NT(Ks + jt * 16 * 16, 16, n, 0, hi, 16), s);
#pragma unroll
      for (int r = 0; r < 8; ++r)
        Pw[(r + 8 * hi) * 256 + jt * 16 + n] = (h16)s[r];
    }
    __builtin_amdgcn_wave_barrier();
    // softmax: 2 lanes per row, 128 columns each, vectorized v8h LDS traffic
    {
      int row = lane >> 1, hsel = lane & 1;
      h16* prow = Pw + row * 256 + hsel * 128;
      const v8h* pv = (const v8h*)prow;
      float mx = -1e30f;
#pragma unroll 4
      for (int j = 0; j < 16; ++j) {
        v8h vj = pv[j];
#pragma unroll
        for (int e = 0; e < 8; ++e) mx = fmaxf(mx, (float)vj[e]);
      }
      mx = fmaxf(mx, __shfl_xor(mx, 1));
      float sum = 0.f;
      v8h* pw = (v8h*)prow;
#pragma unroll 4
      for (int j = 0; j < 16; ++j) {
        v8h vj = pw[j], ej;
#pragma unroll
        for (int e = 0; e < 8; ++e) {
          float ex = __expf((float)vj[e] - mx);
          sum += ex;
          ej[e] = (h16)ex;
        }
        pw[j] = ej;
      }
      sum += __shfl_xor(sum, 1);
      if (hsel == 0) rsc[w][row] = __builtin_amdgcn_rcpf(sum);
    }
    __builtin_amdgcn_wave_barrier();
    // O = P V  (K = 256 -> 8 chained WMMAs; B from transposed Vt, contiguous)
    v8f o = {};
    const h16* parow = Pw + (lane & 15) * 256;
#pragma unroll
    for (int kc = 0; kc < 8; ++kc)
      o = wmma16(mkA(parow, kc * 32, hi, 256), mkB_NT(Vt, 256, n, kc * 32, hi, 256), o);
    // epilogue: normalize rows, apply gate, store f16
#pragma unroll
    for (int r = 0; r < 8; ++r) {
      int m = r + 8 * hi;
      size_t idx = (base + i0 + m) * 64 + h * 16 + n;
      float v = o[r] * rsc[w][m];
      Ph[idx] = (h16)(v * (float)Gh[idx]);
    }
    __builtin_amdgcn_wave_barrier();
  }
}

// ---------------- K3: output projection + bias (fp32 out) ----------------
// block = 256 threads (8 waves), 64 rows per block.
__global__ __launch_bounds__(256) void k_outproj(
    const h16* __restrict__ Ph, const h16* __restrict__ Woh, const float* __restrict__ bo,
    float* __restrict__ out) {
  __shared__ h16 As[64 * 64];  // 8 KB activations
  __shared__ h16 Ws[64 * 64];  // 8 KB Wo
  int t = threadIdx.x;
  size_t row0 = (size_t)blockIdx.x * 64;
  for (int i = t; i < 2048; i += 256) {
    ((u32*)As)[i] = ((const u32*)(Ph + row0 * 64))[i];
    ((u32*)Ws)[i] = ((const u32*)Woh)[i];
  }
  __syncthreads();
  int w = t >> 5, lane = t & 31, n = lane & 15, hi = lane >> 4;
#pragma unroll
  for (int tt = 0; tt < 2; ++tt) {
    int tid = w * 2 + tt;
    int rowtile = tid >> 2, n0 = (tid & 3) * 16;
    const h16* arow = As + (rowtile * 16 + (lane & 15)) * 64;
    v8f c = {};
    c = wmma16(mkA(arow, 0, hi, 64), mkB_NT(Ws, 64, n0 + n, 0, hi, 64), c);
    c = wmma16(mkA(arow, 32, hi, 64), mkB_NT(Ws, 64, n0 + n, 32, hi, 64), c);
    float bias = bo[n0 + n];
#pragma unroll
    for (int r = 0; r < 8; ++r) {
      int m = r + 8 * hi;
      out[(row0 + rowtile * 16 + m) * 64 + n0 + n] = c[r] + bias;
    }
  }
}

// ---------------- host launch ----------------
extern "C" void kernel_launch(void* const* d_in, const int* in_sizes, int n_in,
                              void* d_out, int out_size, void* d_ws, size_t ws_size,
                              hipStream_t stream) {
  (void)in_sizes; (void)n_in; (void)out_size; (void)ws_size;
  const float* z        = (const float*)d_in[0];
  const float* ln_scale = (const float*)d_in[1];
  const float* ln_bias  = (const float*)d_in[2];
  const float* Wq = (const float*)d_in[3];
  const float* Wk = (const float*)d_in[4];
  const float* Wv = (const float*)d_in[5];
  const float* Wg = (const float*)d_in[6];
  const float* bg = (const float*)d_in[7];
  const float* Wo = (const float*)d_in[8];
  const float* bo = (const float*)d_in[9];
  float* out = (float*)d_out;

  // workspace layout (f16 elements)
  h16* wsh = (h16*)d_ws;
  h16* Wqh = wsh;             // [0,     4096)
  h16* Wkh = wsh + 4096;
  h16* Wvh = wsh + 8192;
  h16* Wgh = wsh + 12288;
  h16* Woh = wsh + 16384;
  h16* Qh  = wsh + 32768;     // 5 arrays of RD f16 each (~16 MB apiece)
  h16* Kh  = Qh + RD;
  h16* Vh  = Kh + RD;
  h16* Gh  = Vh + RD;
  h16* Ph  = Gh + RD;

  k_cvt_f16<<<16, 256, 0, stream>>>(Wq, Wqh, 4096);
  k_cvt_f16<<<16, 256, 0, stream>>>(Wk, Wkh, 4096);
  k_cvt_f16<<<16, 256, 0, stream>>>(Wv, Wvh, 4096);
  k_cvt_f16<<<16, 256, 0, stream>>>(Wg, Wgh, 4096);
  k_cvt_f16<<<16, 256, 0, stream>>>(Wo, Woh, 4096);

  // 131072 rows / 32 rows per block
  k_ln_qkvg<<<4096, 256, 0, stream>>>(z, ln_scale, ln_bias, Wqh, bg, Qh, Kh, Vh, Gh);
  // one block per (b, l, h): grid.x = L*H, grid.y = B
  k_attn<<<dim3(LSEQ * 4, NB), 128, 0, stream>>>(Qh, Kh, Vh, Gh, Ph);
  // 131072 rows / 64 rows per block
  k_outproj<<<2048, 256, 0, stream>>>(Ph, Woh, bo, out);
}
